// GraphContrastiveLearningHeads_51788715655459
// MI455X (gfx1250) — matrix operand order repaired and checked by reading, compiled, not run on hardware
//
#include <hip/hip_runtime.h>
#include <hip/hip_bf16.h>

// ---------------------------------------------------------------------------
// Contrastive loss: F = normalize(cat(bert, graph)) [8192x1024]
//   loss = mean_i( log( sum_{j!=i} exp(<F_i,F_j>/T) ) - <F_i, F_{(i+B)%N}>/T )
// |<F_i,F_j>|/T <= 14.3 -> exp never overflows fp32 -> no max tracking.
// GEMM on v_wmma_f32_16x16x32_bf16, fp32 accumulation.
// Each wave computes a 32x16 C strip (two M tiles sharing one B tile) to
// halve global B traffic per WMMA; A panel (32 rows) staged in dynamic LDS.
// ---------------------------------------------------------------------------

typedef __attribute__((ext_vector_type(16))) __bf16 v16bf;
typedef __attribute__((ext_vector_type(8)))  float  v8f;

#define B_ROWS     4096
#define N_ROWS     8192
#define D_DIM      1024
#define INV_T      (1.0f / 0.07f)
#define LDS_STRIDE 1032   // 1024 + 8 halfs pad (row stride 516 dwords -> bank spread)
#define LDS_BYTES  (32 * LDS_STRIDE * 2)   // 66048 B dynamic LDS (<= 320KB/WG on CDNA5)

// ---- round-to-nearest-even fp32 -> bf16 (inputs are finite) ---------------
__device__ __forceinline__ unsigned short f2bf(float f) {
    union { float f; unsigned u; } x; x.f = f;
    unsigned u = x.u;
    unsigned r = u + 0x7FFFu + ((u >> 16) & 1u);
    return (unsigned short)(r >> 16);
}

// ---- zero the per-row accumulators (sum-of-exp and positive logit) --------
__global__ void gcl_init_kernel(float* acc) {
    int i = blockIdx.x * blockDim.x + threadIdx.x;
    if (i < 2 * N_ROWS) acc[i] = 0.0f;
}

// ---- L2-normalize each row of cat(bert, graph), emit bf16 row-major -------
__global__ __launch_bounds__(256)
void gcl_normalize_kernel(const float* __restrict__ graph,
                          const float* __restrict__ bert,
                          unsigned short* __restrict__ Fb) {
    const int row = blockIdx.x;                         // 0..8191
    const float* src = (row < B_ROWS) ? (bert + (size_t)row * D_DIM)
                                      : (graph + (size_t)(row - B_ROWS) * D_DIM);
    const int tid = threadIdx.x;                        // 256 threads, 4 elems each
    float4 v = ((const float4*)src)[tid];
    float ss = v.x * v.x + v.y * v.y + v.z * v.z + v.w * v.w;

    __shared__ float red[256];
    red[tid] = ss;
    __syncthreads();
    for (int s = 128; s > 0; s >>= 1) {
        if (tid < s) red[tid] += red[tid + s];
        __syncthreads();
    }
    const float inv = 1.0f / fmaxf(sqrtf(red[0]), 1e-12f);

    unsigned short* dst = Fb + (size_t)row * D_DIM + tid * 4;
    dst[0] = f2bf(v.x * inv);
    dst[1] = f2bf(v.y * inv);
    dst[2] = f2bf(v.z * inv);
    dst[3] = f2bf(v.w * inv);
}

// ---- fused GEMM (bf16 WMMA) + exp-accumulate + positive capture -----------
// grid.x = 256 row-tiles(32 rows) * 16 col-segments; block = 128 thr = 4 waves.
// Wave: 32 rows x 128 cols = 8 C-tile pairs; K = 1024 (2 wmma per k-step).
__global__ __launch_bounds__(128)
void gcl_simloss_kernel(const unsigned short* __restrict__ Fb,
                        float* __restrict__ sumArr,
                        float* __restrict__ posArr) {
    const int rowTile  = blockIdx.x >> 4;     // 0..255 (32 rows each)
    const int colSeg   = blockIdx.x & 15;     // 0..15  (512 cols each)
    const int tid      = threadIdx.x;
    const int lane     = tid & 31;
    const int wave     = tid >> 5;
    const int rowBase  = rowTile * 32;
    const int half     = lane >> 4;           // 0: lanes 0-15, 1: lanes 16-31
    const int l16      = lane & 15;

    extern __shared__ __align__(16) unsigned short Alds[];  // 32 x LDS_STRIDE halfs

    // Stage the 32-row A panel (32 x 1024 bf16 = 64KB) into LDS, padded rows.
    for (int c = tid; c < 32 * 128; c += 128) {             // 16B chunks
        const int r  = c >> 7;
        const int ch = c & 127;
        const uint4* src = (const uint4*)(Fb + (size_t)(rowBase + r) * D_DIM + ch * 8);
        *(uint4*)(&Alds[r * LDS_STRIDE + ch * 8]) = *src;
    }
    __syncthreads();

    // ISA 16-bit A layout: lanes 0-15 hold K {0..7,16..23}, lanes 16-31 {8..15,24..31}
    const unsigned short* Aptr0 = &Alds[l16 * LDS_STRIDE + half * 8];          // rows 0-15
    const unsigned short* Aptr1 = Aptr0 + 16 * LDS_STRIDE;                     // rows 16-31

    float sumExp[16], posAcc[16];
    #pragma unroll
    for (int v = 0; v < 16; ++v) { sumExp[v] = 0.0f; posAcc[v] = 0.0f; }

    const int colStart = colSeg * 512 + wave * 128;

    for (int ct = 0; ct < 8; ++ct) {
        const int colBase = colStart + ct * 16;
        // B tile = F^T: lane n reads two 16-byte chunks of F row (colBase+n).
        // Lanes 0-15 cover K {0..7,16..23}, lanes 16-31 {8..15,24..31}.
        const unsigned short* Brow =
            Fb + (size_t)(colBase + l16) * D_DIM + half * 8;
        if (ct + 1 < 8)  // warm L2/WGP$ for the next column block
            __builtin_prefetch(Fb + (size_t)(colBase + 16 + l16) * D_DIM, 0, 1);

        v8f acc0 = {0, 0, 0, 0, 0, 0, 0, 0};   // rows rowBase+0..15
        v8f acc1 = {0, 0, 0, 0, 0, 0, 0, 0};   // rows rowBase+16..31
        #pragma unroll 2
        for (int k = 0; k < D_DIM; k += 32) {
            union { v16bf h; uint4 u[2]; } a0, a1, b;
            b.u[0]  = *(const uint4*)(Brow + k);         // K = k+sel   .. +7
            b.u[1]  = *(const uint4*)(Brow + k + 16);    // K = k+sel+16.. +23
            a0.u[0] = *(const uint4*)(Aptr0 + k);        // K = k+sel   .. +7
            a0.u[1] = *(const uint4*)(Aptr0 + k + 16);   // K = k+sel+16.. +23
            a1.u[0] = *(const uint4*)(Aptr1 + k);
            a1.u[1] = *(const uint4*)(Aptr1 + k + 16);
            acc0 = __builtin_amdgcn_wmma_f32_16x16x32_bf16(
                false, a0.h, false, b.h, (short)0, acc0, false, false);
            acc1 = __builtin_amdgcn_wmma_f32_16x16x32_bf16(
                false, a1.h, false, b.h, (short)0, acc1, false, false);
        }

        // C layout: lane l, vgpr v -> (row = v + 8*half, col = l16)
        const int gcol = colBase + l16;
        #pragma unroll
        for (int v = 0; v < 8; ++v) {
            const int grow0 = rowBase + v + half * 8;        // tile 0
            const int grow1 = grow0 + 16;                    // tile 1
            const float s0 = acc0[v] * INV_T;
            const float s1 = acc1[v] * INV_T;
            sumExp[v]     += (gcol == grow0) ? 0.0f : __expf(s0);  // mask diag
            sumExp[v + 8] += (gcol == grow1) ? 0.0f : __expf(s1);
            if (gcol == ((grow0 + B_ROWS) & (N_ROWS - 1))) posAcc[v]     += s0;
            if (gcol == ((grow1 + B_ROWS) & (N_ROWS - 1))) posAcc[v + 8] += s1;
        }
    }

    // Reduce across the 16 lanes that share each row (wave32, width-16 xor).
    #pragma unroll
    for (int v = 0; v < 16; ++v) {
        float se = sumExp[v], pp = posAcc[v];
        for (int m = 8; m >= 1; m >>= 1) {
            se += __shfl_xor(se, m, 16);
            pp += __shfl_xor(pp, m, 16);
        }
        sumExp[v] = se; posAcc[v] = pp;
    }
    if (l16 == 0) {
        #pragma unroll
        for (int v = 0; v < 8; ++v) {
            const int grow0 = rowBase + v + half * 8;
            const int grow1 = grow0 + 16;
            atomicAdd(&sumArr[grow0], sumExp[v]);      // global_atomic_add_f32
            atomicAdd(&posArr[grow0], posAcc[v]);
            atomicAdd(&sumArr[grow1], sumExp[v + 8]);
            atomicAdd(&posArr[grow1], posAcc[v + 8]);
        }
    }
}

// ---- loss = mean( log(S_i) - pos_i ) --------------------------------------
__global__ __launch_bounds__(256)
void gcl_finalize_kernel(const float* __restrict__ sumArr,
                         const float* __restrict__ posArr,
                         float* __restrict__ out) {
    __shared__ float red[256];
    const int tid = threadIdx.x;
    float acc = 0.0f;
    for (int i = tid; i < N_ROWS; i += 256)
        acc += logf(sumArr[i]) - posArr[i];
    red[tid] = acc;
    __syncthreads();
    for (int s = 128; s > 0; s >>= 1) {
        if (tid < s) red[tid] += red[tid + s];
        __syncthreads();
    }
    if (tid == 0) out[0] = red[0] * (1.0f / (float)N_ROWS);
}

// ---------------------------------------------------------------------------
extern "C" void kernel_launch(void* const* d_in, const int* in_sizes, int n_in,
                              void* d_out, int out_size, void* d_ws, size_t ws_size,
                              hipStream_t stream) {
    const float* graph = (const float*)d_in[0];   // graph_pooled [4096,1024] fp32
    const float* bert  = (const float*)d_in[1];   // bert_pooled  [4096,1024] fp32

    // Workspace layout: bf16 F [8192*1024] (16MB) | sumArr[8192] | posArr[8192]
    unsigned short* Fb = (unsigned short*)d_ws;
    float* sumArr = (float*)((char*)d_ws + (size_t)N_ROWS * D_DIM * sizeof(unsigned short));
    float* posArr = sumArr + N_ROWS;
    float* out    = (float*)d_out;

    gcl_init_kernel<<<(2 * N_ROWS + 255) / 256, 256, 0, stream>>>(sumArr);
    gcl_normalize_kernel<<<N_ROWS, 256, 0, stream>>>(graph, bert, Fb);
    gcl_simloss_kernel<<<256 * 16, 128, LDS_BYTES, stream>>>(Fb, sumArr, posArr);
    gcl_finalize_kernel<<<1, 256, 0, stream>>>(sumArr, posArr, out);
}
